// HMM_89017492177135
// MI455X (gfx1250) — matrix operand order repaired and checked by reading, not compiled, not run on hardware
//
#include <hip/hip_runtime.h>
#include <hip/hip_bf16.h>

// HMM forward as a log-sum-exp GEMM scan on CDNA5 WMMA.
//
//   alpha_t[b,j] = m_b + log( sum_i exp(alpha_{t-1}[b,i]-m_b) * exp(A[i,j]) )
//                      + B_emit[j, obs[b,t]]
//
// P = exp(alpha-m) in [0,1], E = exp(A) in (1,e): both f16-safe, f32 accum.
// 4 persistent workgroups x 16 batches (M=16 fills a WMMA tile); the per-step
// global dependency (per-batch max) stays inside one workgroup -> LDS barriers.
// E16 (512 KB) lives in L2 and is re-read each step; an opaque uniform offset
// keeps the compiler from hoisting+spilling the B fragments while preserving
// global address-space provenance (global_load_b128, not flat_load_b128).

typedef __attribute__((ext_vector_type(16))) _Float16 v16h;
typedef __attribute__((ext_vector_type(8)))  _Float16 v8h;
typedef __attribute__((ext_vector_type(8)))  float    v8f;

constexpr int K_    = 512;
constexpr int V_    = 32000;
constexpr int BATCH = 64;
constexpr int T_    = 2048;
constexpr int NT    = K_ / 16;  // 32 n-tiles of 16 states
constexpr int KT    = K_ / 32;  // 16 k-tiles of 32 states
constexpr int BPG   = 16;       // batches per workgroup (== WMMA M)

// ---------------------------------------------------------------------------
// Prep: E16[frag] = (f16) exp(A[i][j]) in this code's WMMA B-operand per-lane order (contiguous K).
// Fragment f = kt*NT + nt holds the 32x16 tile; within a fragment:
//   half index = lane*16 + h, with  N = nt*16 + (lane%16),
//                                   K = kt*32 + (lane/16)*16 + h
// so the hot loop loads v16h at (f*512 + lane*16): coalesced 32B/lane.
// ---------------------------------------------------------------------------
__global__ void hmm_prep_kernel(const float* __restrict__ A,
                                _Float16* __restrict__ E16) {
  int idx = blockIdx.x * blockDim.x + threadIdx.x;  // 0 .. 512*512-1
  int h   = idx & 15;
  int L   = (idx >> 4) & 31;
  int f   = idx >> 9;
  int kt  = f / NT, nt = f % NT;
  int i   = kt * 32 + (L >> 4) * 16 + h;  // row of A (reduction index)
  int j   = nt * 16 + (L & 15);           // state / column of A
  E16[idx] = (_Float16)__expf(A[i * K_ + j]);
}

// ---------------------------------------------------------------------------
// Main scan. grid = BATCH/BPG = 4 blocks, 512 threads = 16 waves (wave32).
// Wave w: (phase 1) max+exp for batch w; (phase 2) n-tiles 2w, 2w+1.
// ---------------------------------------------------------------------------
__global__ void __launch_bounds__(512, 1)
hmm_fwd_kernel(const _Float16* __restrict__ E16,
               const float*    __restrict__ Bemit,
               const float*    __restrict__ pi,
               const int*      __restrict__ obs,
               float*          __restrict__ out) {
  // Row pads: alpha stride 516 dwords and p stride 520 halfs keep the
  // 16-row-strided ds_load_b128 A-fragment reads bank-conflict free.
  __shared__ float    sh_alpha[BPG][K_ + 4];
  __shared__ _Float16 sh_p[BPG][K_ + 8];
  __shared__ float    sh_m[BPG];
  __shared__ int      sh_obs[BPG];

  const int tid  = threadIdx.x;
  const int wave = tid >> 5;
  const int lane = tid & 31;
  const int wg16 = blockIdx.x * BPG;   // first global batch of this workgroup

  // ---- t = 0 : alpha0[b][j] = log(pi[j]) + B_emit[j, obs[b,0]] ----
  if (tid < BPG) sh_obs[tid] = obs[(size_t)(wg16 + tid) * T_];
  __syncthreads();
  {
    int j = tid;  // 512 threads == K states
    float lp = __logf(pi[j]);
#pragma unroll 4
    for (int b = 0; b < BPG; ++b) {
      float v = lp + Bemit[(size_t)j * V_ + sh_obs[b]];
      sh_alpha[b][j] = v;
      out[((size_t)((wg16 + b) * K_ + j)) * T_ + 0] = v;
    }
  }
  __syncthreads();

  // Per-lane WMMA geometry (constant across the scan):
  const int mrow = lane & 15;              // A-frag row M / B-frag column N
  const int ksel = (lane < 16) ? 0 : 16;   // A-frag K phase per lane half
  const int nt0  = wave * 2;               // this wave's first n-tile
  const int j0   = nt0 * 16 + mrow;        // output state of acc0 column
  const int j1   = j0 + 16;                // output state of acc1 column
  const int bhalf = (lane >> 4) * 8;       // C layout: M = r + 8*(lane/16)
  const _Float16* prow = &sh_p[mrow][0];

  // One-time L0/L2 warm-up of this wave's E fragments (global_prefetch_b8);
  // after this the per-step B-fragment loads hit cache.
#pragma unroll
  for (int kt = 0; kt < KT; ++kt) {
    __builtin_prefetch(E16 + ((size_t)(kt * NT + nt0) << 9) + lane * 16, 0, 3);
    __builtin_prefetch(E16 + ((size_t)(kt * NT + nt0 + 1) << 9) + lane * 16, 0, 3);
  }

  for (int t = 1; t < T_; ++t) {
    // ---------- phase 1: m_b = max_j alpha[b][j];  P = exp(alpha - m) -------
    {
      const float* arow = &sh_alpha[wave][0];
      float mx = -3.4e38f;
#pragma unroll
      for (int k = lane; k < K_; k += 32) mx = fmaxf(mx, arow[k]);
#pragma unroll
      for (int off = 16; off > 0; off >>= 1)
        mx = fmaxf(mx, __shfl_xor(mx, off, 32));
      if (lane == 0) sh_m[wave] = mx;
      _Float16* prw = &sh_p[wave][0];
#pragma unroll
      for (int k = lane; k < K_; k += 32)
        prw[k] = (_Float16)__expf(arow[k] - mx);
      if (tid < BPG) sh_obs[tid] = obs[(size_t)(wg16 + tid) * T_ + t];
    }
    __syncthreads();

    // ---------- phase 2: acc[16x16] = P[16x512] @ E[512x16] via WMMA --------
    // Opaque zero offset in an SGPR: the asm redefines it every iteration so
    // the 32 B-fragment loads can't be hoisted (-> no 1KB/lane scratch spill),
    // but the pointer stays rooted at the E16 kernarg so address-space
    // inference keeps the loads as global_load_b128 (LOADcnt only, no DScnt
    // coupling with the LDS A-fragment loads).
    unsigned eoff = 0;
    asm volatile("" : "+s"(eoff));
    const _Float16* eb = E16 + lane * 16 + eoff;

    v8f acc0 = {}, acc1 = {};
#pragma unroll
    for (int kt = 0; kt < KT; ++kt) {
      // A fragment (contiguous K = 16*(lane/16) + s, same order as E16): lane half selects K phase.
      v8h alo = *(const v8h*)(prow + kt * 32 + ksel);        // K = ksel+0..7
      v8h ahi = *(const v8h*)(prow + kt * 32 + ksel + 8);    // K = ksel+8..15
      v16h a  = __builtin_shufflevector(alo, ahi,
                  0, 1, 2, 3, 4, 5, 6, 7, 8, 9, 10, 11, 12, 13, 14, 15);
      // B fragments: fragment-major E16, 32B/lane fully coalesced.
      const _Float16* f0 = eb + ((size_t)(kt * NT + nt0) << 9);
      v16h b0 = *(const v16h*)(f0);
      v16h b1 = *(const v16h*)(f0 + 512);
      acc0 = __builtin_amdgcn_wmma_f32_16x16x32_f16(
                 false, a, false, b0, (short)0, acc0, false, false);
      acc1 = __builtin_amdgcn_wmma_f32_16x16x32_f16(
                 false, a, false, b1, (short)0, acc1, false, false);
    }

    // ---------- phase 3: alpha_new = m + log(acc) + emission ----------------
    // C layout: VGPR r, lanes 0-15 -> M=r, lanes 16-31 -> M=r+8; N = lane%16.
#pragma unroll
    for (int r = 0; r < 8; ++r) {
      int   b = r + bhalf;
      float m = sh_m[b];
      int   o = sh_obs[b];
      float a0 = m + __logf(acc0[r]) + Bemit[(size_t)j0 * V_ + o];
      float a1 = m + __logf(acc1[r]) + Bemit[(size_t)j1 * V_ + o];
      sh_alpha[b][j0] = a0;
      sh_alpha[b][j1] = a1;
      out[((size_t)((wg16 + b) * K_ + j0)) * T_ + t] = a0;
      out[((size_t)((wg16 + b) * K_ + j1)) * T_ + t] = a1;
    }
    __syncthreads();
  }
}

extern "C" void kernel_launch(void* const* d_in, const int* in_sizes, int n_in,
                              void* d_out, int out_size, void* d_ws, size_t ws_size,
                              hipStream_t stream) {
  (void)in_sizes; (void)n_in; (void)out_size; (void)ws_size;
  const float* A     = (const float*)d_in[0];   // [K, K]
  const float* Bemit = (const float*)d_in[1];   // [K, V]
  const float* pi    = (const float*)d_in[2];   // [K]
  const int*   obs   = (const int*)d_in[3];     // [BATCH, T]
  float*       out   = (float*)d_out;           // [BATCH, K, T]
  _Float16*    E16   = (_Float16*)d_ws;         // 512 KB fragment-major exp(A)

  hmm_prep_kernel<<<dim3(K_ * K_ / 512), dim3(512), 0, stream>>>(A, E16);
  hmm_fwd_kernel<<<dim3(BATCH / BPG), dim3(512), 0, stream>>>(E16, Bemit, pi, obs, out);
}
